// WiderAttnDecompClassCoherence_38285338477244
// MI455X (gfx1250) — compile-verified, matrix operand order checked by reading, not run on hardware
//
#include <hip/hip_runtime.h>
#include <math.h>

// ---------------------------------------------------------------------------
// MI455X (gfx1250) implementation.
// Dominant work: 3x 65-tap conv == 65 accumulated (4096x512)@(512x32) GEMM taps
// -> f16 WMMA (v_wmma_f32_16x16x32_f16), wave32, 16x32 C-tile per wave.
// Conv tap weights staged global->LDS with GLOBAL_LOAD_ASYNC_TO_LDS_B128
// (ASYNCcnt double-buffer pipeline), consumed as ds_load_b128 B-fragments.
// ---------------------------------------------------------------------------

typedef __attribute__((ext_vector_type(16))) _Float16 v16h;
typedef __attribute__((ext_vector_type(8)))  float    v8f;

#define B_   4
#define L_   1024
#define NF_  480
#define H_   32
#define FC_  512
#define D_   512
#define W_   65
#define ROWS (B_ * L_)   // 4096
#define KT_  (FC_ / 32)  // 16 k-steps of 32
#define PKS  16384       // packed elems for one 512x32 B matrix (32 KB)

#define WMMA_F16(A, Bf, C) \
  __builtin_amdgcn_wmma_f32_16x16x32_f16(false, (A), false, (Bf), (short)0, (C), false, false)

// K index inside a 16-bit A/B fragment: element e (0..15), lane-half (0/1).
__device__ __forceinline__ int kmap(int e, int half) {
  return (e < 8) ? (8 * half + e) : (16 + 8 * half + (e - 8));
}

// Load one 16x32 f16 A fragment from a row-major matrix (ld elements/row).
// Per lane: two contiguous 16B chunks. `valid=false` -> zero fragment (padding).
__device__ __forceinline__ v16h load_a_rowmajor(const _Float16* __restrict__ A,
                                                long row, int ld, int kb,
                                                int half, bool valid) {
  v16h a;
  if (valid) {
    const _Float16* p = A + row * (long)ld + kb + 8 * half;
#pragma unroll
    for (int i = 0; i < 8; ++i) { a[i] = p[i]; a[8 + i] = p[16 + i]; }
  } else {
#pragma unroll
    for (int i = 0; i < 16; ++i) a[i] = (_Float16)0.f;
  }
  return a;
}

// ---------------- P1: softmax + concat -> oef (f16), and m = mask + collect --
__global__ void __launch_bounds__(256)
prep_oef_kernel(const float* __restrict__ xe, const float* __restrict__ lat,
                const float* __restrict__ mask, const float* __restrict__ x,
                _Float16* __restrict__ oef, float* __restrict__ mvec) {
  int row  = blockIdx.x * (blockDim.x >> 5) + (threadIdx.x >> 5);
  int lane = threadIdx.x & 31;
  if (row >= ROWS) return;
  float mk = mask[row];
  float v  = lat[row * H_ + lane] * mk;
  float mx = v;
#pragma unroll
  for (int s = 16; s > 0; s >>= 1) mx = fmaxf(mx, __shfl_xor(mx, s, 32));
  float e = __expf(v - mx);
  float sum = e;
#pragma unroll
  for (int s = 16; s > 0; s >>= 1) sum += __shfl_xor(sum, s, 32);
  oef[(size_t)row * FC_ + NF_ + lane] = (_Float16)((e / sum) * mk);
  for (int c = lane; c < NF_; c += 32)
    oef[(size_t)row * FC_ + c] = (_Float16)xe[(size_t)row * NF_ + c];
  if (lane == 0) mvec[row] = mk + x[row * 23 + 21] + x[row * 23 + 22];
}

// ---------------- P2: pack knn (2 x 512x512) into B-fragment order ----------
__global__ void pack_knn_kernel(const float* __restrict__ knn,
                                _Float16* __restrict__ knnpk) {
  int idx = blockIdx.x * blockDim.x + threadIdx.x;
  if (idx >= 2 * FC_ * FC_) return;
  int i = idx >> 18, r = idx & (262144 - 1);
  int e = r & 15, lane = (r >> 4) & 31, kt = (r >> 9) & 15, nt = r >> 13;
  int k = kt * 32 + kmap(e, lane >> 4);
  int n = nt * 16 + (lane & 15);
  knnpk[idx] = (_Float16)knn[((size_t)i * FC_ + k) * FC_ + n];
}

// ---------------- P3: build + pack the 3 Gaussian conv kernels (65 taps) ----
__global__ void pack_ktaps_kernel(const float* __restrict__ ampl,
                                  const float* __restrict__ inp,
                                  const float* __restrict__ outv,
                                  const float* __restrict__ pos_embed,
                                  _Float16* __restrict__ kpk) {
  int idx = blockIdx.x * blockDim.x + threadIdx.x;
  if (idx >= 3 * W_ * PKS) return;
  int r = idx & (PKS - 1);
  int w = (idx >> 14) % W_;
  int c = idx / (W_ * PKS);
  int e = r & 15, lane = (r >> 4) & 31, kt = (r >> 9) & 15, nt = (r >> 13) & 1;
  int d = kt * 32 + kmap(e, lane >> 4);
  int h = nt * 16 + (lane & 15);
  float pos = pos_embed[w];
  float mean, ivar, am;
  if (c == 0) {
    mean = inp[0 * D_ + d] * outv[0 * H_ + h];
    ivar = fabsf(inp[1 * D_ + d] * outv[1 * H_ + h]) + 0.01f;
    am   = ampl[(0 * D_ + d) * H_ + h];
  } else if (c == 1) {
    mean = 0.f;
    ivar = fabsf(inp[2 * D_ + d] * outv[2 * H_ + h]) + 0.01f;
    am   = ampl[(1 * D_ + d) * H_ + h];
  } else {
    mean = inp[3 * D_ + d] * outv[3 * H_ + h];
    ivar = fabsf(inp[4 * D_ + d] * outv[4 * H_ + h]) + 0.01f;
    am   = ampl[(2 * D_ + d) * H_ + h];
  }
  float dd = pos - mean;
  kpk[idx] = (_Float16)(__expf(-ivar * dd * dd) * am);
}

// ---------------- P4: pack small 512x32 weight matrices ---------------------
__global__ void pack_small_kernel(const float* __restrict__ ampl,
                                  const float* __restrict__ posw,
                                  _Float16* __restrict__ spk) {
  int idx = blockIdx.x * blockDim.x + threadIdx.x;
  if (idx >= 4 * PKS) return;
  int r = idx & (PKS - 1), which = idx >> 14;
  int e = r & 15, lane = (r >> 4) & 31, kt = (r >> 9) & 15, nt = (r >> 13) & 1;
  int k = kt * 32 + kmap(e, lane >> 4);
  int n = nt * 16 + (lane & 15);
  float v;
  if (which == 0)      v = ampl[(1 * D_ + k) * H_ + n];  // A[1]
  else if (which == 1) v = ampl[(3 * D_ + k) * H_ + n];  // A[3]
  else if (which == 2) v = ampl[(4 * D_ + k) * H_ + n];  // A[4]
  else                 v = posw[k * H_ + n];             // pos_dep weights
  spk[idx] = (_Float16)v;
}

// ---------------- G1: cf = oef @ knn[i]  (4096x512)@(512x512), f16 WMMA -----
__global__ void __launch_bounds__(256)
gemm_cf_kernel(const _Float16* __restrict__ oef, const _Float16* __restrict__ knnpk,
               _Float16* __restrict__ cf0, _Float16* __restrict__ cf1) {
  int wave = threadIdx.x >> 5, lane = threadIdx.x & 31;
  int t = blockIdx.x * 8 + wave;  // 8192 wavetiles
  int which = blockIdx.y;
  int mt = t >> 5, nt = t & 31;
  const _Float16* bpk = knnpk + (size_t)which * FC_ * FC_;
  _Float16* cf = which ? cf1 : cf0;
  int half = lane >> 4;
  long row = mt * 16 + (lane & 15);
  v8f acc = {};
#pragma unroll 4
  for (int kt = 0; kt < KT_; ++kt) {
    v16h a = load_a_rowmajor(oef, row, FC_, kt * 32, half, true);
    v16h b = *(const v16h*)(bpk + (((size_t)(nt * KT_ + kt) * 32 + lane) << 4));
    acc = WMMA_F16(a, b, acc);
  }
  int n = nt * 16 + (lane & 15);
#pragma unroll
  for (int v = 0; v < 8; ++v)
    cf[(size_t)(mt * 16 + v + 8 * half) * FC_ + n] = (_Float16)acc[v];
}

// ---------------- G2: three 65-tap convs as accumulated WMMA GEMMs ----------
// Tap weights (32 KB each) staged to LDS with async global->LDS DMA,
// double-buffered across taps; B fragments then come from ds_load_b128.
__global__ void __launch_bounds__(256)
conv_wmma_kernel(const _Float16* __restrict__ cf0, const _Float16* __restrict__ cf1,
                 const _Float16* __restrict__ kpk, float* __restrict__ acts) {
  __shared__ _Float16 kbuf[2][PKS];  // 2 x 32 KB double buffer
  int tid  = threadIdx.x;
  int wave = tid >> 5, lane = tid & 31;
  int mt = blockIdx.x * 8 + wave;  // 256 M-tiles
  int c  = blockIdx.y;             // conv 0/1/2
  const _Float16* cf = (c == 1) ? cf1 : cf0;
  const _Float16* kc = kpk + (size_t)c * W_ * PKS;
  float* actc = acts + (size_t)c * ROWS * H_;
  int half = lane >> 4;
  int b = mt >> 6;                         // batch
  int l = ((mt & 63) << 4) + (lane & 15);  // row within batch

  // LDS byte addresses (generic-pointer low 32 bits == LDS offset).
  unsigned lds_base[2];
  lds_base[0] = (unsigned)(size_t)(&kbuf[0][0]);
  lds_base[1] = (unsigned)(size_t)(&kbuf[1][0]);
  unsigned my_off = (unsigned)tid * 16u;  // per-thread 16B slice, stride 4 KB

  // Prologue: async-stage tap 0 into buffer 0 (8 x b128 per thread).
  {
    const char* src = (const char*)kc + my_off;
    unsigned dst = lds_base[0] + my_off;
#pragma unroll
    for (int i = 0; i < 8; ++i)
      asm volatile("global_load_async_to_lds_b128 %0, %1, off"
                   :: "v"(dst + (unsigned)(i * 4096)),
                      "v"(src + (size_t)(i * 4096))
                   : "memory");
  }

  v8f acc0 = {}, acc1 = {};
#pragma unroll 1
  for (int w = 0; w < W_; ++w) {
    if (w + 1 < W_) {
      // Kick off tap w+1 into the other buffer, then wait for tap w only.
      unsigned dst = lds_base[(w + 1) & 1] + my_off;
      const char* src = (const char*)(kc + (size_t)(w + 1) * PKS) + my_off;
#pragma unroll
      for (int i = 0; i < 8; ++i)
        asm volatile("global_load_async_to_lds_b128 %0, %1, off"
                     :: "v"(dst + (unsigned)(i * 4096)),
                        "v"(src + (size_t)(i * 4096))
                     : "memory");
      asm volatile("s_wait_asynccnt 0x8" ::: "memory");
    } else {
      asm volatile("s_wait_asynccnt 0x0" ::: "memory");
    }
    __syncthreads();  // all waves' slices of tap w are in LDS

    const _Float16* kw = &kbuf[w & 1][0];
    int lr = l + w - 32;
    bool valid = (lr >= 0) && (lr < L_);
    long grow = (long)b * L_ + lr;
#pragma unroll 4
    for (int kt = 0; kt < KT_; ++kt) {
      v16h a  = load_a_rowmajor(cf, grow, FC_, kt * 32, half, valid);
      v16h f0 = *(const v16h*)(kw + ((kt * 32 + lane) << 4));
      acc0 = WMMA_F16(a, f0, acc0);
      v16h f1 = *(const v16h*)(kw + (((KT_ + kt) * 32 + lane) << 4));
      acc1 = WMMA_F16(a, f1, acc1);
    }
    __syncthreads();  // buffer w&1 free for reuse at tap w+2
  }
#pragma unroll
  for (int v = 0; v < 8; ++v) {
    int mrow = mt * 16 + v + 8 * half;
    actc[(size_t)mrow * H_ + (lane & 15)]      = acc0[v];
    actc[(size_t)mrow * H_ + 16 + (lane & 15)] = acc1[v];
  }
}

// ---------------- G3: four (4096x512)@(512x32) small GEMMs ------------------
__global__ void __launch_bounds__(256)
gemm_small_kernel(const _Float16* __restrict__ oef, const _Float16* __restrict__ cf0,
                  const _Float16* __restrict__ cf1, const _Float16* __restrict__ spk,
                  float* __restrict__ outs) {
  int wave = threadIdx.x >> 5, lane = threadIdx.x & 31;
  int mt = blockIdx.x * 8 + wave;
  int which = blockIdx.y;  // 0: cf1@A1  1: cf0@A3  2: cf1@A4  3: oef@Wpd
  const _Float16* A = (which == 1) ? cf0 : (which == 3 ? oef : cf1);
  const _Float16* bpk = spk + (size_t)which * PKS;
  float* out = outs + (size_t)which * ROWS * H_;
  int half = lane >> 4;
  long row = mt * 16 + (lane & 15);
  v8f acc0 = {}, acc1 = {};
#pragma unroll 4
  for (int kt = 0; kt < KT_; ++kt) {
    v16h a  = load_a_rowmajor(A, row, FC_, kt * 32, half, true);
    v16h f0 = *(const v16h*)(bpk + ((kt * 32 + lane) << 4));
    acc0 = WMMA_F16(a, f0, acc0);
    v16h f1 = *(const v16h*)(bpk + (((KT_ + kt) * 32 + lane) << 4));
    acc1 = WMMA_F16(a, f1, acc1);
  }
#pragma unroll
  for (int v = 0; v < 8; ++v) {
    int mrow = mt * 16 + v + 8 * half;
    out[(size_t)mrow * H_ + (lane & 15)]      = acc0[v];
    out[(size_t)mrow * H_ + 16 + (lane & 15)] = acc1[v];
  }
}

// ---------------- V: Gaussian Gram contraction, m row staged in LDS ---------
__global__ void __launch_bounds__(256)
vcorr_kernel(const float* __restrict__ mvec, const float* __restrict__ bias,
             float* __restrict__ Vc) {
  __shared__ float ms[L_];
  int b  = blockIdx.x >> 7;
  int i0 = (blockIdx.x & 127) << 3;
  for (int j = threadIdx.x; j < L_; j += 256) ms[j] = mvec[b * L_ + j];
  __syncthreads();
  int h = threadIdx.x & 31;
  int i = i0 + (threadIdx.x >> 5);
  float cm  = bias[4 * H_ + h];
  float sig = cm * cm + 1e-4f;
  float fi = (float)i, s = 0.f;
  for (int j = 0; j < L_; ++j) {
    float d = fi - (float)j;
    s += __expf(-sig * d * d) * ms[j];
  }
  Vc[((size_t)b * L_ + i) * H_ + h] = ms[i] * s;
}

// ---------------- F: elementwise combine ------------------------------------
__global__ void final_kernel(const float* __restrict__ acts, const float* __restrict__ souts,
                             const float* __restrict__ Vc, const float* __restrict__ bias,
                             const float* __restrict__ mask, float* __restrict__ out) {
  int idx = blockIdx.x * blockDim.x + threadIdx.x;
  if (idx >= ROWS * H_) return;
  int h = idx & 31, row = idx >> 5;
  float a0  = acts[idx];
  float a1  = acts[ROWS * H_ + idx];
  float a2  = acts[2 * ROWS * H_ + idx];
  float a1b = souts[idx];
  float p3v = souts[ROWS * H_ + idx];
  float p4v = souts[2 * ROWS * H_ + idx];
  float pp  = souts[3 * ROWS * H_ + idx];
  float b0 = bias[h], b1v = bias[H_ + h], b2 = bias[2 * H_ + h], b3 = bias[3 * H_ + h];
  float rdf = (a0 + b0) * a1 + a2 * (a1b + b2) + (p3v + b1v) * p4v;
  out[idx] = Vc[idx] * (pp + b3) * rdf * mask[row];
}

// ---------------------------------------------------------------------------
extern "C" void kernel_launch(void* const* d_in, const int* in_sizes, int n_in,
                              void* d_out, int out_size, void* d_ws, size_t ws_size,
                              hipStream_t stream) {
  const float* xe   = (const float*)d_in[0];   // (4,1024,480)
  const float* lat  = (const float*)d_in[1];   // (4,1024,32)
  const float* mask = (const float*)d_in[2];   // (4,1024,1)
  const float* pos  = (const float*)d_in[3];   // (65,1,1)
  const float* x    = (const float*)d_in[4];   // (4,1024,23)
  const float* ampl = (const float*)d_in[5];   // (5,512,32)
  const float* inp  = (const float*)d_in[6];   // (5,512,1)
  const float* outv = (const float*)d_in[7];   // (5,1,32)
  const float* bias = (const float*)d_in[8];   // (5,1,32)
  const float* pdw  = (const float*)d_in[9];   // (1,512,32)
  const float* knn  = (const float*)d_in[10];  // (2,512,512)
  float* out = (float*)d_out;

  char* ws = (char*)d_ws;
  size_t off = 0;
  auto alloc = [&](size_t bytes) -> char* {
    char* p = ws + off;
    off = (off + bytes + 255) & ~(size_t)255;
    return p;
  };
  _Float16* oef   = (_Float16*)alloc((size_t)ROWS * FC_ * 2);
  _Float16* cf0   = (_Float16*)alloc((size_t)ROWS * FC_ * 2);
  _Float16* cf1   = (_Float16*)alloc((size_t)ROWS * FC_ * 2);
  _Float16* knnpk = (_Float16*)alloc((size_t)2 * FC_ * FC_ * 2);
  _Float16* kpk   = (_Float16*)alloc((size_t)3 * W_ * PKS * 2);
  _Float16* spk   = (_Float16*)alloc((size_t)4 * PKS * 2);
  float* acts  = (float*)alloc((size_t)3 * ROWS * H_ * 4);
  float* souts = (float*)alloc((size_t)4 * ROWS * H_ * 4);
  float* Vc    = (float*)alloc((size_t)ROWS * H_ * 4);
  float* mvec  = (float*)alloc((size_t)ROWS * 4);

  prep_oef_kernel<<<ROWS / 8, 256, 0, stream>>>(xe, lat, mask, x, oef, mvec);
  pack_knn_kernel<<<(2 * FC_ * FC_ + 255) / 256, 256, 0, stream>>>(knn, knnpk);
  pack_ktaps_kernel<<<(3 * W_ * PKS + 255) / 256, 256, 0, stream>>>(ampl, inp, outv, pos, kpk);
  pack_small_kernel<<<(4 * PKS + 255) / 256, 256, 0, stream>>>(ampl, pdw, spk);
  gemm_cf_kernel<<<dim3(1024, 2), 256, 0, stream>>>(oef, knnpk, cf0, cf1);
  conv_wmma_kernel<<<dim3(32, 3), 256, 0, stream>>>(cf0, cf1, kpk, acts);
  gemm_small_kernel<<<dim3(32, 4), 256, 0, stream>>>(oef, cf0, cf1, spk, souts);
  vcorr_kernel<<<512, 256, 0, stream>>>(mvec, bias, Vc);
  final_kernel<<<(ROWS * H_ + 255) / 256, 256, 0, stream>>>(acts, souts, Vc, bias, mask, out);
}
